// CoeffNet_62234076119442
// MI455X (gfx1250) — matrix operand order repaired and checked by reading, not compiled, not run on hardware
//
#include <hip/hip_runtime.h>

// ---------------------------------------------------------------------------
// CoeffNet (equivariant attention) for gfx1250 / MI455X, wave32 WMMA pipeline.
// All matmul-shaped stages run v_wmma_f32_16x16x32_f16 with f16 operands and
// f32 accumulation; intermediates are f16 and stay L2-resident (~150MB << 192MB).
// Workspace requirement: ~144.2 MB.
// ---------------------------------------------------------------------------

typedef __attribute__((ext_vector_type(16))) _Float16 v16h;
typedef __attribute__((ext_vector_type(8)))  _Float16 v8h;
typedef __attribute__((ext_vector_type(8)))  float    v8f;

#define B_DIM   4
#define A_DIM   64
#define NSITE   16384          // B*A*A
#define F_IN    32
#define F_DIM   64
#define L_FULL  9
#define PL      18             // 2 parities * 9 lm
#define ROWLEN  1152           // PL * F_DIM
#define KTOT    73728          // A_DIM * ROWLEN

struct CGnz { int m, n, k; float v; };

__constant__ int DMAPc[9] = {0,1,1,1,2,2,2,2,2};

// ---- WMMA fragment helpers (CDNA5 16x16x32 f16 layouts) -------------------
__device__ inline v16h cat8(v8h lo, v8h hi) {
  return __builtin_shufflevector(lo, hi, 0,1,2,3,4,5,6,7,8,9,10,11,12,13,14,15);
}
// A operand 16x32 f16, k order chosen here (B loads match): lane half h, row lane%16. element e -> K=h*16+e.
// One contiguous 32B load per lane.
__device__ inline v16h load_a_f16(const _Float16* row, int k0, int hlf) {
  return *(const v16h*)(row + k0 + hlf * 16);
}
__device__ inline v8f wmma32(v16h a, v16h b, v8f c) {
  return __builtin_amdgcn_wmma_f32_16x16x32_f16(false, a, false, b, (short)0, c,
                                                false, false);
}

// ---- Clebsch-Gordan init (once per call, tiny) ----------------------------
__device__ double dfact(int n) {
  double r = 1.0;
  for (int i = 2; i <= n; ++i) r *= (double)i;
  return r;
}

__device__ double cg_complex(int j1,int m1,int j2,int m2,int j3,int m3) {
  if (m1 + m2 != m3) return 0.0;
  int lo = j1 - j2; if (lo < 0) lo = -lo;
  if (j3 < lo || j3 > j1 + j2) return 0.0;
  double pre = sqrt((double)(2*j3+1) * dfact(j1+j2-j3) * dfact(j1-j2+j3) *
                    dfact(-j1+j2+j3) / dfact(j1+j2+j3+1));
  pre *= sqrt(dfact(j1+m1)*dfact(j1-m1)*dfact(j2+m2)*dfact(j2-m2)*
              dfact(j3+m3)*dfact(j3-m3));
  double s = 0.0;
  for (int k = 0; k <= j1+j2+j3; ++k) {
    int d1 = j1+j2-j3-k, d2 = j1-m1-k, d3 = j2+m2-k;
    int d4 = j3-j2+m1+k, d5 = j3-j1-m2+k;
    if (d1<0 || d2<0 || d3<0 || d4<0 || d5<0) continue;
    double den = dfact(k)*dfact(d1)*dfact(d2)*dfact(d3)*dfact(d4)*dfact(d5);
    s += ((k & 1) ? -1.0 : 1.0) / den;
  }
  return pre * s;
}

__device__ void u_real_d(int l, double Ur[5][5], double Ui[5][5]) {
  for (int i = 0; i < 5; ++i)
    for (int j = 0; j < 5; ++j) { Ur[i][j] = 0.0; Ui[i][j] = 0.0; }
  Ur[l][l] = 1.0;
  const double is2 = 0.70710678118654752440;
  for (int m = 1; m <= l; ++m) {
    double sgn = (m & 1) ? -1.0 : 1.0;
    Ur[l+m][l+m] = sgn * is2;
    Ur[l+m][l-m] = is2;
    Ui[l-m][l-m] = is2;
    Ui[l-m][l+m] = -sgn * is2;
  }
}

__global__ void init_cg_kernel(CGnz* __restrict__ nz, int* __restrict__ nnz_p) {
  __shared__ float Gs[729];
  const int t = threadIdx.x;
  if (t < 27) {
    const int l1 = t / 9, l2 = (t / 3) % 3, l3 = t % 3;
    const int n1 = 2*l1+1, n2 = 2*l2+1, n3 = 2*l3+1;
    double Cc[5][5][5];
    for (int i = 0; i < n1; ++i)
      for (int j = 0; j < n2; ++j)
        for (int k = 0; k < n3; ++k)
          Cc[i][j][k] = cg_complex(l1, i-l1, l2, j-l2, l3, k-l3);
    double U1r[5][5], U1i[5][5], U2r[5][5], U2i[5][5], U3r[5][5], U3i[5][5];
    u_real_d(l1, U1r, U1i); u_real_d(l2, U2r, U2i); u_real_d(l3, U3r, U3i);
    for (int a = 0; a < n1; ++a)
      for (int b = 0; b < n2; ++b)
        for (int c = 0; c < n3; ++c) {
          double gr = 0.0, gi = 0.0;
          for (int i = 0; i < n1; ++i)
            for (int j = 0; j < n2; ++j) {
              const double u1r = U1r[a][i], u1i = U1i[a][i];
              const double u2r = U2r[b][j], u2i = U2i[b][j];
              const double pr = u1r*u2r - u1i*u2i;
              const double pi = u1r*u2i + u1i*u2r;
              for (int k = 0; k < n3; ++k) {
                const double cv = Cc[i][j][k];
                if (cv == 0.0) continue;
                const double u3r = U3r[c][k], u3i = U3i[c][k];
                gr += (pr*u3r + pi*u3i) * cv;   // * conj(U3)
                gi += (pi*u3r - pr*u3i) * cv;
              }
            }
          const double g = (((l1+l2+l3) & 1) == 0) ? gr : gi;
          Gs[(l1*l1 + a)*81 + (l2*l2 + b)*9 + (l3*l3 + c)] = (float)g;
        }
  }
  __syncthreads();
  if (t == 0) {
    int cnt = 0;
    for (int m = 0; m < 9; ++m)
      for (int n = 0; n < 9; ++n)
        for (int k = 0; k < 9; ++k) {
          const float g = Gs[(m*9 + n)*9 + k];
          if (fabsf(g) > 1e-7f) { nz[cnt].m = m; nz[cnt].n = n; nz[cnt].k = k;
                                  nz[cnt].v = g; ++cnt; }
        }
    *nnz_p = cnt;
  }
}

// ---- equiv_dense, F_IN=32 -> 64, f32 input, dual kernels (k1/k2) ----------
// grid (1024 row-tiles, 18 pm, 2 which), block 32 (one wave)
__global__ void proj32_kernel(const float* __restrict__ x,
                              const float* __restrict__ k1,
                              const float* __restrict__ k2,
                              _Float16* __restrict__ o1,
                              _Float16* __restrict__ o2) {
  const int lane = threadIdx.x;
  const int hlf = lane >> 4, l16 = lane & 15;
  const int tile = blockIdx.x;
  const int pm = blockIdx.y;
  const int p = pm / L_FULL, m = pm % L_FULL;
  const int dm = DMAPc[m];
  const float*    w   = blockIdx.z ? k2 : k1;
  _Float16*       out = blockIdx.z ? o2 : o1;
  const int site = tile*16 + l16;
  const float* xrow = x + ((size_t)(site*2 + p)*L_FULL + m) * F_IN;
  v8h lo, hi;
  #pragma unroll
  for (int e = 0; e < 8; ++e) {
    lo[e] = (_Float16)xrow[hlf*16 + e];
    hi[e] = (_Float16)xrow[hlf*16 + 8 + e];
  }
  const v16h afrag = cat8(lo, hi);
  const float* wbase = w + (size_t)(p*3 + dm) * F_IN * F_DIM;
  #pragma unroll
  for (int nt = 0; nt < 4; ++nt) {
    const int col = nt*16 + l16;
    v16h bfrag;
    #pragma unroll
    for (int e = 0; e < 16; ++e)
      bfrag[e] = (_Float16)wbase[(size_t)(hlf*16 + e) * F_DIM + col];
    v8f c = {};
    c = wmma32(afrag, bfrag, c);
    #pragma unroll
    for (int v = 0; v < 8; ++v) {
      const int row = tile*16 + v + 8*hlf;
      out[((size_t)(row*2 + p)*L_FULL + m) * F_DIM + col] = (_Float16)c[v];
    }
  }
}

// ---- equiv_dense, 64 -> 64, f16 input (Q/K/V projections) -----------------
// grid (1024, 18), block 32
__global__ void proj64_kernel(const _Float16* __restrict__ h,
                              const float* __restrict__ w,
                              _Float16* __restrict__ out) {
  const int lane = threadIdx.x;
  const int hlf = lane >> 4, l16 = lane & 15;
  const int tile = blockIdx.x;
  const int pm = blockIdx.y;
  const int p = pm / L_FULL, m = pm % L_FULL;
  const int dm = DMAPc[m];
  const int site = tile*16 + l16;
  const _Float16* hrow = h + ((size_t)(site*2 + p)*L_FULL + m) * F_DIM;
  const float* wbase = w + (size_t)(p*3 + dm) * F_DIM * F_DIM;
  const v16h a0 = load_a_f16(hrow, 0, hlf);
  const v16h a1 = load_a_f16(hrow, 32, hlf);
  #pragma unroll
  for (int nt = 0; nt < 4; ++nt) {
    const int col = nt*16 + l16;
    v16h b0, b1;
    #pragma unroll
    for (int e = 0; e < 16; ++e) {
      b0[e] = (_Float16)wbase[(size_t)(hlf*16 + e) * F_DIM + col];
      b1[e] = (_Float16)wbase[(size_t)(32 + hlf*16 + e) * F_DIM + col];
    }
    v8f c = {};
    c = wmma32(a0, b0, c);
    c = wmma32(a1, b1, c);
    #pragma unroll
    for (int v = 0; v < 8; ++v) {
      const int row = tile*16 + v + 8*hlf;
      out[((size_t)(row*2 + p)*L_FULL + m) * F_DIM + col] = (_Float16)c[v];
    }
  }
}

// ---- CG bilinear (tensor_dense core): h[c,k,f] = sum G*w*x1*x2 ------------
// block (64 f, 4 sites), grid 4096
__global__ void bilinear_kernel(const _Float16* __restrict__ x1,
                                const _Float16* __restrict__ x2,
                                const float* __restrict__ w,
                                const CGnz* __restrict__ nz,
                                const int* __restrict__ nnz_p,
                                _Float16* __restrict__ out) {
  __shared__ float sh[4][64][56];   // 18 acc + 18 v1 + 18 v2 per thread
  const int f = threadIdx.x;
  const int ty = threadIdx.y;
  const int site = blockIdx.x*4 + ty;
  float* acc = &sh[ty][f][0];
  float* v1  = &sh[ty][f][18];
  float* v2  = &sh[ty][f][36];
  const _Float16* b1 = x1 + (size_t)site*ROWLEN + f;
  const _Float16* b2 = x2 + (size_t)site*ROWLEN + f;
  #pragma unroll
  for (int i = 0; i < PL; ++i) {
    acc[i] = 0.f;
    v1[i] = (float)b1[(size_t)i * F_DIM];
    v2[i] = (float)b2[(size_t)i * F_DIM];
  }
  const int nnz = *nnz_p;
  for (int i = 0; i < nnz; ++i) {
    const CGnz e = nz[i];
    const int dm = DMAPc[e.m], dn = DMAPc[e.n], dk = DMAPc[e.k];
    const float gv = e.v;
    #pragma unroll
    for (int a = 0; a < 2; ++a) {
      #pragma unroll
      for (int c = 0; c < 2; ++c) {
        const int bb = a ^ c;
        const float wv = w[(size_t)((((a*2 + bb)*3 + dm)*3 + dn)*3 + dk) * F_DIM + f];
        acc[c*L_FULL + e.k] += gv * wv * v1[a*L_FULL + e.m] * v2[bb*L_FULL + e.n];
      }
    }
  }
  #pragma unroll
  for (int i = 0; i < PL; ++i)
    out[(size_t)site*ROWLEN + (size_t)i*F_DIM + f] = (_Float16)acc[i];
}

// ---- attention score init: dot = mask (broadcast over batch) --------------
__global__ void dot_init_kernel(const float* __restrict__ mask,
                                float* __restrict__ dot) {
  const int i = blockIdx.x*256 + threadIdx.x;   // 16384
  dot[i] = mask[i & 4095];
}

// ---- dot[B,a,c] += Q[B,a,:] . K[B,c,:], K-split GEMM via WMMA -------------
// grid (36 ksplits, 16 tiles, 4 batches), block 32
__global__ void dot_accum_kernel(const _Float16* __restrict__ Q,
                                 const _Float16* __restrict__ Kb,
                                 float* __restrict__ dot) {
  const int lane = threadIdx.x;
  const int hlf = lane >> 4, l16 = lane & 15;
  const int ks = blockIdx.x;
  const int tile = blockIdx.y;
  const int Bb = blockIdx.z;
  const int ta = tile >> 2, tc = tile & 3;
  const _Float16* qrow = Q + (size_t)(Bb*4096 + (ta*16 + l16)*A_DIM) * ROWLEN;
  const _Float16* krow = Kb + (size_t)(Bb*4096 + (tc*16 + l16)*A_DIM) * ROWLEN;
  v8f c = {};
  const int k0 = ks * 2048;
  for (int kk = 0; kk < 2048; kk += 32) {
    __builtin_prefetch(qrow + k0 + kk + 512, 0, 1);
    const v16h a = load_a_f16(qrow, k0 + kk, hlf);
    const v16h b = *(const v16h*)(krow + k0 + kk + hlf*16);   // B = K^T: contiguous
    c = wmma32(a, b, c);
  }
  #pragma unroll
  for (int v = 0; v < 8; ++v) {
    const int ai = ta*16 + v + 8*hlf;
    const int ci = tc*16 + l16;
    atomicAdd(&dot[Bb*4096 + ai*A_DIM + ci], c[v]);
  }
}

// ---- softmax over 'a' for each (B,c) column -------------------------------
__global__ void softmax_kernel(const float* __restrict__ dot,
                               _Float16* __restrict__ w16) {
  const int bc = blockIdx.x;            // 256 = B*64
  const int Bb = bc >> 6, cc = bc & 63;
  const int a = threadIdx.x;            // 64
  __shared__ float sh[64];
  const float v = dot[Bb*4096 + a*64 + cc];
  sh[a] = v;
  __syncthreads();
  for (int s = 32; s > 0; s >>= 1) {
    if (a < s) sh[a] = fmaxf(sh[a], sh[a + s]);
    __syncthreads();
  }
  const float mx = sh[0];
  __syncthreads();
  const float e = __expf(v - mx);
  sh[a] = e;
  __syncthreads();
  for (int s = 32; s > 0; s >>= 1) {
    if (a < s) sh[a] += sh[a + s];
    __syncthreads();
  }
  w16[Bb*4096 + a*64 + cc] = (_Float16)(e / sh[0]);
}

// ---- out[B,a,n] = sum_b w[B,a,b] * V[B,b,n], n = c*1152+plf ---------------
// grid (4608 n-tiles, 4 a-tiles, 4 batches), block 32
__global__ void apply_kernel(const _Float16* __restrict__ w16,
                             const _Float16* __restrict__ V,
                             _Float16* __restrict__ out) {
  const int lane = threadIdx.x;
  const int hlf = lane >> 4, l16 = lane & 15;
  const int nt = blockIdx.x;
  const int ta = blockIdx.y;
  const int Bb = blockIdx.z;
  const _Float16* wrow = w16 + (size_t)Bb*4096 + (size_t)(ta*16 + l16)*A_DIM;
  const int ncol = nt*16 + l16;
  const _Float16* vbase = V + (size_t)Bb*4096*ROWLEN;
  _Float16* obase = out + (size_t)Bb*4096*ROWLEN;
  v8f c = {};
  #pragma unroll
  for (int kstep = 0; kstep < 64; kstep += 32) {
    const v16h a = load_a_f16(wrow, kstep, hlf);
    v16h b;
    #pragma unroll
    for (int e = 0; e < 16; ++e) {
      const int bi = kstep + hlf*16 + e;
      b[e] = vbase[(size_t)bi*KTOT + ncol];
    }
    c = wmma32(a, b, c);
  }
  #pragma unroll
  for (int v = 0; v < 8; ++v) {
    const int arow = ta*16 + v + 8*hlf;
    obase[(size_t)arow*KTOT + ncol] = (_Float16)c[v];
  }
}

// ---- tdo + final reduction to 4 outputs -----------------------------------
__global__ void zero_out_kernel(float* __restrict__ out) {
  if (threadIdx.x < B_DIM) out[threadIdx.x] = 0.f;
}

__global__ void tdo_kernel(const _Float16* __restrict__ h,
                           const float* __restrict__ k1,
                           const float* __restrict__ k2,
                           const float* __restrict__ w,
                           const CGnz* __restrict__ nz,
                           const int* __restrict__ nnz_p,
                           float* __restrict__ out4) {
  const int site = blockIdx.x*256 + threadIdx.x;   // 16384
  float y1[PL], y2[PL];
  #pragma unroll
  for (int i = 0; i < PL; ++i) {
    const int p = i / L_FULL, m = i % L_FULL, dm = DMAPc[m];
    const _Float16* hr = h + (size_t)site*ROWLEN + (size_t)i*F_DIM;
    const float* w1 = k1 + (p*3 + dm)*F_DIM;
    const float* w2 = k2 + (p*3 + dm)*F_DIM;
    float s1 = 0.f, s2 = 0.f;
    for (int f = 0; f < F_DIM; ++f) {
      const float hv = (float)hr[f];
      s1 += hv * w1[f];
      s2 += hv * w2[f];
    }
    y1[i] = s1; y2[i] = s2;
  }
  float acc = 0.f;
  const int nnz = *nnz_p;
  for (int i = 0; i < nnz; ++i) {
    const CGnz e = nz[i];
    if (e.k != 0) continue;             // out_lm == 1
    const int dm = DMAPc[e.m], dn = DMAPc[e.n];
    #pragma unroll
    for (int a = 0; a < 2; ++a) {       // include_pseudo=False -> c=0, b=a
      const float wv = w[((a*2 + a)*3 + dm)*3 + dn];
      acc += e.v * wv * y1[a*L_FULL + e.m] * y2[a*L_FULL + e.n];
    }
  }
  atomicAdd(&out4[site >> 12], acc);    // site/4096 = batch
}

// ---------------------------------------------------------------------------
extern "C" void kernel_launch(void* const* d_in, const int* in_sizes, int n_in,
                              void* d_out, int out_size, void* d_ws, size_t ws_size,
                              hipStream_t stream) {
  const float* x      = (const float*)d_in[0];
  const float* mask   = (const float*)d_in[1];
  const float* td1_k1 = (const float*)d_in[2];
  const float* td1_k2 = (const float*)d_in[3];
  const float* td1_w  = (const float*)d_in[4];
  const float* h1_qk  = (const float*)d_in[5];
  const float* h1_kk  = (const float*)d_in[6];
  const float* h1_vk  = (const float*)d_in[7];
  const float* h2_qk  = (const float*)d_in[8];
  const float* h2_kk  = (const float*)d_in[9];
  const float* h2_vk  = (const float*)d_in[10];
  const float* tdo_k1 = (const float*)d_in[11];
  const float* tdo_k2 = (const float*)d_in[12];
  const float* tdo_w  = (const float*)d_in[13];
  float* out = (float*)d_out;

  char* ws = (char*)d_ws;
  int*      nnz    = (int*)ws;
  CGnz*     nzlist = (CGnz*)(ws + 256);
  _Float16* w16    = (_Float16*)(ws + 16384);      // 4*4096 halves
  float*    dotb   = (float*)(ws + 49152);         // 4*4096 floats
  const size_t HB = (size_t)NSITE * ROWLEN * sizeof(_Float16);  // 37.75MB
  _Float16* bufA = (_Float16*)(ws + 131072);
  _Float16* bufB = (_Float16*)(ws + 131072 + HB);
  _Float16* bufC = (_Float16*)(ws + 131072 + 2*HB);
  _Float16* bufD = (_Float16*)(ws + 131072 + 3*HB);

  // 1) Clebsch-Gordan coefficients -> compact nonzero list
  init_cg_kernel<<<1, 32, 0, stream>>>(nzlist, nnz);

  // 2) td1: x1 = x@k1, x2 = x@k2 (WMMA), then CG bilinear -> h in bufC
  proj32_kernel<<<dim3(1024, 18, 2), 32, 0, stream>>>(x, td1_k1, td1_k2, bufA, bufB);
  bilinear_kernel<<<dim3(4096), dim3(64, 4), 0, stream>>>(bufA, bufB, td1_w,
                                                          nzlist, nnz, bufC);

  // 3) two attention heads
  for (int head = 0; head < 2; ++head) {
    const float* qk = head ? h2_qk : h1_qk;
    const float* kk = head ? h2_kk : h1_kk;
    const float* vk = head ? h2_vk : h1_vk;
    proj64_kernel<<<dim3(1024, 18), 32, 0, stream>>>(bufC, qk, bufA);  // Q
    proj64_kernel<<<dim3(1024, 18), 32, 0, stream>>>(bufC, kk, bufB);  // K
    proj64_kernel<<<dim3(1024, 18), 32, 0, stream>>>(bufC, vk, bufD);  // V
    dot_init_kernel<<<dim3(64), 256, 0, stream>>>(mask, dotb);
    dot_accum_kernel<<<dim3(36, 16, 4), 32, 0, stream>>>(bufA, bufB, dotb);
    softmax_kernel<<<dim3(256), 64, 0, stream>>>(dotb, w16);
    apply_kernel<<<dim3(4608, 4, 4), 32, 0, stream>>>(w16, bufD, bufC);
  }

  // 4) tdo readout + per-batch reduction
  zero_out_kernel<<<1, 32, 0, stream>>>(out);
  tdo_kernel<<<dim3(64), 256, 0, stream>>>(bufC, tdo_k1, tdo_k2, tdo_w,
                                           nzlist, nnz, out);

  (void)in_sizes; (void)n_in; (void)out_size; (void)ws_size;
}
